// FusedMoe_4587025072789
// MI455X (gfx1250) — compile-verified, matrix operand order checked by reading, not run on hardware
//
#include <hip/hip_runtime.h>
#include <hip/hip_bf16.h>

// ---------------- problem constants (match reference) ----------------
enum { E_EXP = 16, M_TOK = 4096, N_FFN = 1408, K_HID = 2048, TOPK = 2 };
enum { ROWS_TOTAL = M_TOK * TOPK };           // 8192 expert-sorted row slots
#define LSTR 72   // LDS row stride in halves (144B): 16B-aligned chunks, conflict-free

typedef __attribute__((ext_vector_type(16))) _Float16 v16h;
typedef __attribute__((ext_vector_type(8)))  _Float16 v8h;
typedef __attribute__((ext_vector_type(4)))  _Float16 v4h;
typedef __attribute__((ext_vector_type(8)))  float    v8f;

// ---------------- small utility kernels ----------------
__global__ void moe_zero_f32(float* __restrict__ p, int n) {
    int i = blockIdx.x * blockDim.x + threadIdx.x;
    if (i < n) p[i] = 0.0f;
}

__global__ void moe_zero_i32(int* __restrict__ p, int n) {
    int i = blockIdx.x * blockDim.x + threadIdx.x;
    if (i < n) p[i] = 0;
}

__global__ void moe_count(const int* __restrict__ ids, int* __restrict__ counts, int n) {
    int i = blockIdx.x * blockDim.x + threadIdx.x;
    if (i < n) atomicAdd(&counts[ids[i]], 1);
}

__global__ void moe_scan(const int* __restrict__ counts,
                         int* __restrict__ offsets, int* __restrict__ cursor) {
    if (threadIdx.x == 0 && blockIdx.x == 0) {
        int s = 0;
        for (int e = 0; e < E_EXP; ++e) {
            offsets[e] = s;
            cursor[e]  = s;
            s += counts[e];
        }
        offsets[E_EXP] = s;   // == ROWS_TOTAL
    }
}

__global__ void moe_scatter(const int* __restrict__ ids, const float* __restrict__ tkw,
                            int* __restrict__ cursor, int* __restrict__ tok,
                            float* __restrict__ tws, int n) {
    int i = blockIdx.x * blockDim.x + threadIdx.x;
    if (i < n) {
        int e = ids[i];
        int p = atomicAdd(&cursor[e], 1);
        tok[p] = i / TOPK;          // token index
        tws[p] = tkw[i];            // routing weight for this slot
    }
}

// ---------------- helpers ----------------
__device__ __forceinline__ v4h cvt4(float4 x) {
    v4h p; p[0] = (_Float16)x.x; p[1] = (_Float16)x.y;
           p[2] = (_Float16)x.z; p[3] = (_Float16)x.w; return p;
}

__device__ __forceinline__ v8h cvt8(float4 x, float4 y) {
    v8h p;
    p[0] = (_Float16)x.x; p[1] = (_Float16)x.y; p[2] = (_Float16)x.z; p[3] = (_Float16)x.w;
    p[4] = (_Float16)y.x; p[5] = (_Float16)y.y; p[6] = (_Float16)y.z; p[7] = (_Float16)y.w;
    return p;
}

// one-time f32 -> f16 pre-convert of the activation matrix (4 elems/thread)
__global__ void moe_cvt_f32_f16(const float* __restrict__ src, _Float16* __restrict__ dst, int n) {
    int i = 4 * (blockIdx.x * blockDim.x + threadIdx.x);
    if (i < n) *(v4h*)(dst + i) = cvt4(*(const float4*)(src + i));
}

// CDNA5 async global->LDS copy (ASYNCcnt-tracked; 16B per lane).
// ISA 15.18.3: GLOBAL_LOAD_ASYNC_TO_LDS_B128  vdst(=LDS byte addr), vaddr, off
__device__ __forceinline__ void async_ld_lds_b128(void* lds_dst, const void* gsrc) {
    unsigned loff = (unsigned)(uintptr_t)lds_dst;   // low 32 bits of flat = LDS byte addr
    asm volatile("global_load_async_to_lds_b128 %0, %1, off"
                 :: "v"(loff), "v"((unsigned long long)(uintptr_t)gsrc)
                 : "memory");
}

__device__ __forceinline__ void wait_async0() {
#if __has_builtin(__builtin_amdgcn_s_wait_asynccnt)
    __builtin_amdgcn_s_wait_asynccnt(0);
#else
    asm volatile("s_wait_asynccnt 0" ::: "memory");
#endif
}

// WMMA fragment from an LDS tile row (lane half h -> K chunks h*8 and 16+h*8).
__device__ __forceinline__ v16h ldfrag(const _Float16* base) {
    v8h x = *(const v8h*)base;
    v8h y = *(const v8h*)(base + 16);
    v16h d;
#pragma unroll
    for (int i = 0; i < 8; ++i) { d[i] = x[i]; d[8 + i] = y[i]; }
    return d;
}

__device__ __forceinline__ v8f wmma_f16(const v16h& a, const v16h& b, const v8f& c) {
    return __builtin_amdgcn_wmma_f32_16x16x32_f16(
        false, a, false, b, (short)0, c, false, false);
}

__device__ __forceinline__ float fast_silu(float g) {
    return g * __builtin_amdgcn_rcpf(1.0f + __expf(-g));   // v_exp_f32 + v_rcp_f32
}

// ---------------- GEMM1 + SiLU*mul (fused gate/up), LDS-staged, BK=64 ----------------
// grid: (N_FFN/64, rowTiles, E); block: 256 threads = 8 waves (2 row x 4 col)
__global__ __launch_bounds__(256) void moe_gemm1_silu(
    const _Float16* __restrict__ hid16, const float* __restrict__ w1,
    const int* __restrict__ offsets, const int* __restrict__ tok,
    _Float16* __restrict__ hbuf) {
    __shared__ _Float16 As [32 * LSTR];   // 32 rows x 64 k (f16)
    __shared__ _Float16 Bgs[64 * LSTR];   // 64 gate cols x 64 k
    __shared__ _Float16 Bus[64 * LSTR];   // 64 up   cols x 64 k

    int e    = blockIdx.z;
    int seg0 = offsets[e];
    int rows = offsets[e + 1] - seg0;
    int rowBase = blockIdx.y * 32;
    if (rowBase >= rows) return;                  // uniform: before any barrier

    int t    = threadIdx.x;
    int wave = t >> 5, lane = t & 31;
    int h = lane >> 4, ml = lane & 15;
    int m0l = (wave >> 2) * 16;                   // wave's local row base (0/16)
    int j0  = blockIdx.x * 64 + (wave & 3) * 16;  // gate column base; up = j0+N

    // staging: A = one async b128 (8 halves)/thread; Bg/Bu = 16 floats/thread
    int arow_l = t >> 3;                          // 0..31
    int ak     = (t & 7) * 8;                     // halves 0..56
    int grow   = rowBase + arow_l; if (grow >= rows) grow = rows - 1;  // clamp
    const _Float16* Ag = hid16 + (size_t)tok[seg0 + grow] * K_HID + ak;
    _Float16*       Al = &As[arow_l * LSTR + ak];
    int brow_l = t >> 2;                          // 0..63
    int bk     = (t & 3) * 16;                    // 0,16,32,48
    const float* Bgg = w1 + ((size_t)e * (2 * N_FFN) + (size_t)(blockIdx.x * 64 + brow_l)) * K_HID + bk;
    const float* Bug = Bgg + (size_t)N_FFN * K_HID;
    _Float16* Bgl = &Bgs[brow_l * LSTR + bk];
    _Float16* Bul = &Bus[brow_l * LSTR + bk];

    v8f cg = {}; v8f cu = {};
    for (int kb = 0; kb < K_HID; kb += 64) {
        async_ld_lds_b128(Al, Ag + kb);           // activations: pure f16 async copy
        {                                         // weights: f32 -> f16 convert-on-stage
            const float4* p = (const float4*)(Bgg + kb);
            *(v8h*)(Bgl)     = cvt8(p[0], p[1]);
            *(v8h*)(Bgl + 8) = cvt8(p[2], p[3]);
            const float4* q = (const float4*)(Bug + kb);
            *(v8h*)(Bul)     = cvt8(q[0], q[1]);
            *(v8h*)(Bul + 8) = cvt8(q[2], q[3]);
        }
        wait_async0();
        __syncthreads();

        const _Float16* ar = &As [(m0l + ml) * LSTR + h * 8];
        const _Float16* gr = &Bgs[((wave & 3) * 16 + ml) * LSTR + h * 8];
        const _Float16* ur = &Bus[((wave & 3) * 16 + ml) * LSTR + h * 8];
        v16h a0 = ldfrag(ar),      a1 = ldfrag(ar + 32);
        v16h g0 = ldfrag(gr),      g1 = ldfrag(gr + 32);
        v16h u0 = ldfrag(ur),      u1 = ldfrag(ur + 32);
        cg = wmma_f16(a0, g0, cg); cg = wmma_f16(a1, g1, cg);
        cu = wmma_f16(a0, u0, cu); cu = wmma_f16(a1, u1, cu);
        __syncthreads();
    }

    // epilogue: h = silu(gate) * up  -> f16 compact buffer [ROWS_TOTAL, N]
    int col = j0 + ml;
#pragma unroll
    for (int r = 0; r < 8; ++r) {
        int row = rowBase + m0l + r + 8 * h;
        if (row < rows) {
            float v = fast_silu(cg[r]) * cu[r];
            hbuf[(size_t)(seg0 + row) * N_FFN + col] = (_Float16)v;
        }
    }
}

// ---------------- GEMM2 + weighted combine, LDS-staged, BN=64 ----------------
// grid: (K_HID/64, rowTiles, E); block: 256 threads = 8 waves
__global__ __launch_bounds__(256) void moe_gemm2_combine(
    const _Float16* __restrict__ hbuf, const float* __restrict__ w2,
    const int* __restrict__ offsets, const int* __restrict__ tok,
    const float* __restrict__ tws, float* __restrict__ out) {
    __shared__ _Float16 As[32 * LSTR];
    __shared__ _Float16 Bs[64 * LSTR];

    int e    = blockIdx.z;
    int seg0 = offsets[e];
    int rows = offsets[e + 1] - seg0;
    int rowBase = blockIdx.y * 32;
    if (rowBase >= rows) return;                  // uniform

    int t    = threadIdx.x;
    int wave = t >> 5, lane = t & 31;
    int h = lane >> 4, ml = lane & 15;
    int m0l = (wave >> 2) * 16;
    int k0  = blockIdx.x * 64 + (wave & 3) * 16;  // output column within K

    int arow_l = t >> 3;
    int an     = (t & 7) * 8;
    int grow   = rowBase + arow_l; if (grow >= rows) grow = rows - 1;
    const _Float16* Ag = hbuf + (size_t)(seg0 + grow) * N_FFN + an;
    _Float16*       Al = &As[arow_l * LSTR + an];
    int brow_l = t >> 2;
    int bn     = (t & 3) * 16;
    const float* Bg = w2 + ((size_t)e * K_HID + (size_t)(blockIdx.x * 64 + brow_l)) * N_FFN + bn;
    _Float16*    Bl = &Bs[brow_l * LSTR + bn];

    v8f c = {};
    for (int nb = 0; nb < N_FFN; nb += 64) {
        async_ld_lds_b128(Al, Ag + nb);           // h tile: already f16, async copy
        {
            const float4* p = (const float4*)(Bg + nb);
            *(v8h*)(Bl)     = cvt8(p[0], p[1]);
            *(v8h*)(Bl + 8) = cvt8(p[2], p[3]);
        }
        wait_async0();
        __syncthreads();

        const _Float16* ar = &As[(m0l + ml) * LSTR + h * 8];
        const _Float16* br = &Bs[((wave & 3) * 16 + ml) * LSTR + h * 8];
        v16h a0 = ldfrag(ar), a1 = ldfrag(ar + 32);
        v16h b0 = ldfrag(br), b1 = ldfrag(br + 32);
        c = wmma_f16(a0, b0, c);
        c = wmma_f16(a1, b1, c);
        __syncthreads();
    }

    // epilogue: out[token, k] += route_w * y   (TOPK slots folded by atomics)
    int k = k0 + ml;
#pragma unroll
    for (int r = 0; r < 8; ++r) {
        int row = rowBase + m0l + r + 8 * h;
        if (row < rows) {
            int   tix = tok[seg0 + row];
            float w   = tws[seg0 + row];
            atomicAdd(&out[(size_t)tix * K_HID + k], w * c[r]);
        }
    }
}

// ---------------- launcher ----------------
extern "C" void kernel_launch(void* const* d_in, const int* in_sizes, int n_in,
                              void* d_out, int out_size, void* d_ws, size_t ws_size,
                              hipStream_t stream) {
    (void)in_sizes; (void)n_in; (void)out_size; (void)ws_size;
    const float* hidden = (const float*)d_in[0];   // [M, K]
    const float* w1     = (const float*)d_in[1];   // [E, 2N, K]
    const float* w2     = (const float*)d_in[2];   // [E, K, N]
    const float* tkw    = (const float*)d_in[3];   // [M, TOPK]
    const int*   tki    = (const int*)d_in[4];     // [M, TOPK]
    float* out = (float*)d_out;                    // [M, K]

    // workspace layout
    int*   counts  = (int*)d_ws;                       // E
    int*   offsets = counts + E_EXP;                   // E+1
    int*   cursor  = offsets + (E_EXP + 1);            // E
    int*   tok     = cursor + E_EXP;                   // ROWS_TOTAL
    float* tws     = (float*)(tok + ROWS_TOTAL);       // ROWS_TOTAL
    _Float16* hid16 = (_Float16*)((char*)d_ws + 65792);             // [M, K] f16
    _Float16* hbuf  = hid16 + (size_t)M_TOK * K_HID;                // [ROWS_TOTAL, N] f16

    // 1) zero output + counters; pre-convert activations to f16
    moe_zero_f32<<<(M_TOK * K_HID + 255) / 256, 256, 0, stream>>>(out, M_TOK * K_HID);
    moe_zero_i32<<<1, 64, 0, stream>>>(counts, E_EXP);
    moe_cvt_f32_f16<<<(M_TOK * K_HID / 4 + 255) / 256, 256, 0, stream>>>(hidden, hid16, M_TOK * K_HID);

    // 2) routing: histogram -> scan -> scatter (expert-sorted row lists)
    moe_count<<<(ROWS_TOTAL + 255) / 256, 256, 0, stream>>>(tki, counts, ROWS_TOTAL);
    moe_scan<<<1, 32, 0, stream>>>(counts, offsets, cursor);
    moe_scatter<<<(ROWS_TOTAL + 255) / 256, 256, 0, stream>>>(tki, tkw, cursor, tok, tws, ROWS_TOTAL);

    // 3) GEMM1 (gate|up) + silu*mul -> hbuf (f16)
    dim3 g1(N_FFN / 64, (ROWS_TOTAL + 31) / 32, E_EXP);
    moe_gemm1_silu<<<g1, 256, 0, stream>>>(hid16, w1, offsets, tok, hbuf);

    // 4) GEMM2 + weighted combine -> out
    dim3 g2(K_HID / 64, (ROWS_TOTAL + 31) / 32, E_EXP);
    moe_gemm2_combine<<<g2, 256, 0, stream>>>(hbuf, w2, offsets, tok, tws, out);
}
